// MultiHeadSelfAttention_63702954934685
// MI455X (gfx1250) — matrix operand order repaired and checked by reading, not compiled, not run on hardware
//
#include <hip/hip_runtime.h>
#include <hip/hip_bf16.h>

typedef __attribute__((ext_vector_type(16))) _Float16 v16h;
typedef __attribute__((ext_vector_type(8)))  _Float16 v8h;
typedef __attribute__((ext_vector_type(4)))  _Float16 v4h;
typedef __attribute__((ext_vector_type(8)))  float    v8f;

typedef unsigned int u32x4 __attribute__((ext_vector_type(4)));
typedef int          i32x8 __attribute__((ext_vector_type(8)));
typedef int          i32x4 __attribute__((ext_vector_type(4)));

#define HEADS 16
#define HDIM  64

#if defined(__HIP_DEVICE_COMPILE__) && __has_builtin(__builtin_amdgcn_tensor_load_to_lds)
#define HAVE_TDM 1
#else
#define HAVE_TDM 0
#endif

// ---------------------------------------------------------------------------
// TDM: issue a 2D tensor DMA (f16 elements) Global -> LDS, compact row-major
// destination. Descriptor layout per CDNA5 ISA.
//   tile: tile_w x tile_h elements, source row stride = stride elems,
//   tensor_dim0 = tile_w, tensor_dim1 = tens_h (for OOB clamp).
// Issued once per wave (EXEC ignored); tracked by TENSORcnt.
// ---------------------------------------------------------------------------
#if HAVE_TDM
__device__ inline void tdm_load_2d_f16(const void* gptr, unsigned lds_byte_addr,
                                       unsigned tile_w, unsigned tile_h,
                                       unsigned stride, unsigned tens_h) {
  unsigned long long ga = (unsigned long long)(uintptr_t)gptr;
  u32x4 g0;
  g0[0] = 1u;                                            // count=1 (valid), user mode
  g0[1] = lds_byte_addr;                                 // lds_addr
  g0[2] = (unsigned)(ga & 0xffffffffu);                  // global_addr[31:0]
  g0[3] = (unsigned)((ga >> 32) & 0x01ffffffu)           // global_addr[56:32]
        | (2u << 30);                                    // type=2 ("image")
  i32x8 g1;
  g1[0] = (int)(1u << 16);                               // data_size=1 (2 bytes)
  g1[1] = (int)(tile_w << 16);                           // tensor_dim0[15:0] (<=64K)
  g1[2] = (int)((tens_h & 0xffffu) << 16);               // dim0 hi=0 | tensor_dim1 lo16
  g1[3] = (int)((tens_h >> 16) & 0xffffu)                // tensor_dim1 hi16
        | (int)(tile_w << 16);                           // tile_dim0
  g1[4] = (int)tile_h;                                   // tile_dim1 (tile_dim2=0)
  g1[5] = (int)stride;                                   // tensor_dim0_stride[31:0]
  g1[6] = 0;                                             // stride hi16 | dim1_stride lo16
  g1[7] = 0;
  i32x4 z4 = {0, 0, 0, 0};
#if __clang_major__ >= 23
  i32x8 z8 = {0, 0, 0, 0, 0, 0, 0, 0};
  __builtin_amdgcn_tensor_load_to_lds(g0, g1, z4, z4, z8, 0);
#else
  __builtin_amdgcn_tensor_load_to_lds(g0, g1, z4, z4, 0);
#endif
}
#endif

// ---------------------------------------------------------------------------
// A-fragment loader: 16x32 f16 A matrix (M x K) from row-major src, row stride ld.
// Layout (matches load_b32x16): lane m<16 holds K=0..15;
// lane m+16 holds K=16..31.
// ---------------------------------------------------------------------------
__device__ inline v16h load_a16x32(const _Float16* base, int row0, int ld, int k0, int lane) {
  int m = lane & 15, hi = (lane >> 4) & 1;
  const _Float16* r = base + (size_t)(row0 + m) * ld + k0 + hi * 16;
  v8h lo = *(const v8h*)(r);
  v8h hh = *(const v8h*)(r + 8);
  v16h out;
#pragma unroll
  for (int i = 0; i < 8; ++i) { out[i] = lo[i]; out[8 + i] = hh[i]; }
  return out;
}

// ---------------------------------------------------------------------------
// B-fragment loader: 32x16 f16 B matrix (K x N) where B[k][n] = src[col0+n][k].
// Lane n<16 holds K=0..15 of column n; lane n+16 holds K=16..31. Contiguous 32B.
// ---------------------------------------------------------------------------
__device__ inline v16h load_b32x16(const _Float16* base, int col0, int ld, int k0, int lane) {
  int n = lane & 15, hi = (lane >> 4) & 1;
  const _Float16* r = base + (size_t)(col0 + n) * ld + k0 + hi * 16;
  return *(const v16h*)(r);
}

// ---------------------------------------------------------------------------
// fp32 -> f16 cast (vector-of-4 per thread)
// ---------------------------------------------------------------------------
__global__ void cast_f32_to_f16(const float* __restrict__ src, _Float16* __restrict__ dst, int n) {
  int i = (blockIdx.x * blockDim.x + threadIdx.x) * 4;
  if (i + 3 < n) {
    float4 v = *(const float4*)(src + i);
    v4h o;
    o[0] = (_Float16)v.x; o[1] = (_Float16)v.y; o[2] = (_Float16)v.z; o[3] = (_Float16)v.w;
    *(v4h*)(dst + i) = o;
  } else {
    for (int j = i; j < n; ++j) dst[j] = (_Float16)src[j];
  }
}

// ---------------------------------------------------------------------------
// GEMM: out[M,N] = A[M,K] @ W[N,K]^T + bias. f16 inputs, f32 accumulate.
// Block = 256 threads (8 waves) for load-latency hiding; each wave computes a
// 32x64 tile (2x4 wmma frags). W fragments come straight from global: with
// 192MB L2 the whole weight matrix is cache-resident.
// ---------------------------------------------------------------------------
__global__ void gemm_bias_wmma(const _Float16* __restrict__ A, const _Float16* __restrict__ W,
                               const float* __restrict__ bias,
                               _Float16* __restrict__ outh, float* __restrict__ outf,
                               int M, int N, int K) {
  int lane = threadIdx.x & 31;
  int wave = threadIdx.x >> 5;
  int row0 = blockIdx.x * 256 + wave * 32;
  int n0   = blockIdx.y * 64;

  v8f acc[2][4];
#pragma unroll
  for (int mt = 0; mt < 2; ++mt)
#pragma unroll
    for (int nt = 0; nt < 4; ++nt) acc[mt][nt] = {};

  for (int k0 = 0; k0 < K; k0 += 32) {
    if (k0 + 64 < K) {  // pull next A K-slice toward the WGP (global_prefetch_b8)
      __builtin_prefetch(A + (size_t)(row0 + (lane & 15)) * K + k0 + 64, 0, 1);
    }
    v16h a0 = load_a16x32(A, row0,      K, k0, lane);
    v16h a1 = load_a16x32(A, row0 + 16, K, k0, lane);
#pragma unroll
    for (int nt = 0; nt < 4; ++nt) {
      v16h b = load_b32x16(W, n0 + nt * 16, K, k0, lane);
      acc[0][nt] = __builtin_amdgcn_wmma_f32_16x16x32_f16(false, a0, false, b, (short)0,
                                                          acc[0][nt], false, false);
      acc[1][nt] = __builtin_amdgcn_wmma_f32_16x16x32_f16(false, a1, false, b, (short)0,
                                                          acc[1][nt], false, false);
    }
  }

  // C/D layout: lane n<16 holds col n rows 0-7 in v0-7; lane n+16 holds rows 8-15.
  int col = lane & 15, rsel = (lane >> 4) & 1;
  if (outf) {
#pragma unroll
    for (int mt = 0; mt < 2; ++mt)
#pragma unroll
      for (int nt = 0; nt < 4; ++nt) {
        int c = n0 + nt * 16 + col;
        float bv = bias[c];
#pragma unroll
        for (int i = 0; i < 8; ++i) {
          int r = row0 + mt * 16 + rsel * 8 + i;
          outf[(size_t)r * N + c] = acc[mt][nt][i] + bv;
        }
      }
  } else {
#pragma unroll
    for (int mt = 0; mt < 2; ++mt)
#pragma unroll
      for (int nt = 0; nt < 4; ++nt) {
        int c = n0 + nt * 16 + col;
        float bv = bias[c];
#pragma unroll
        for (int i = 0; i < 8; ++i) {
          int r = row0 + mt * 16 + rsel * 8 + i;
          outh[(size_t)r * N + c] = (_Float16)(acc[mt][nt][i] + bv);
        }
      }
  }
}

// ---------------------------------------------------------------------------
// Flash attention: one block = 4 waves = 64 query rows of one (b,h).
// K tiles (32x64 f16, row stride D) are staged into LDS by the Tensor Data
// Mover, double-buffered so the next tile's DMA overlaps compute
// (s_wait_tensorcnt 1 with one in-flight, in-order completion). V is staged
// transposed (d-major) by the threads so PV B-fragments are contiguous LDS
// reads. Online softmax uses wave32 shfl_xor butterflies over the 16-lane
// column groups; P goes C-layout -> LDS -> A-layout for the PV wmma.
// ---------------------------------------------------------------------------
__global__ void flash_attn_wmma(const _Float16* __restrict__ Q, const _Float16* __restrict__ Km,
                                const _Float16* __restrict__ V, const int* __restrict__ mask,
                                _Float16* __restrict__ O, int S, int D) {
  int lane = threadIdx.x & 31;
  int wave = threadIdx.x >> 5;
  int bh = blockIdx.y;
  int b = bh / HEADS, h = bh % HEADS;
  int q0 = blockIdx.x * 64 + wave * 16;
  size_t baseBH = (size_t)b * S * D + (size_t)h * HDIM;

  __shared__ _Float16 kbuf[2][32 * HDIM];   // [buf][key][d], TDM destination
  __shared__ _Float16 vbufT[HDIM * 32];     // [d][key]
  __shared__ _Float16 pbuf[4][16 * 32];     // per-wave P tile, row-major 16x32

  // Q fragments (16 rows x 64 d = two K=32 chunks), pre-scaled by 1/sqrt(HD)=0.125
  v16h qf[2];
#pragma unroll
  for (int dk = 0; dk < 2; ++dk) {
    qf[dk] = load_a16x32(Q + baseBH, q0, D, dk * 32, lane);
#pragma unroll
    for (int i = 0; i < 16; ++i) qf[dk][i] = qf[dk][i] * (_Float16)0.125f;
  }

  v8f acc[4];
#pragma unroll
  for (int nt = 0; nt < 4; ++nt) acc[nt] = {};
  float mrun[8], lrun[8];
#pragma unroll
  for (int i = 0; i < 8; ++i) { mrun[i] = -3.0e38f; lrun[i] = 0.0f; }

  int col = lane & 15, rsel = (lane >> 4) & 1;
  int nTiles = S / 32;

#if HAVE_TDM
  if (wave == 0) {  // prologue DMA: first K tile into buffer 0
    tdm_load_2d_f16(Km + baseBH, (unsigned)(uintptr_t)&kbuf[0][0], HDIM, 32, (unsigned)D,
                    (unsigned)S);
  }
#endif

  for (int t = 0; t < nTiles; ++t) {
    int kb = t * 32;
    __syncthreads();  // previous compute done: safe to overwrite vbufT / next kbuf
    {   // stage V transposed: 128 threads x 16 halfs
      int tt = threadIdx.x;
      int krow = tt >> 2, seg = (tt & 3) * 16;
      v16h vv = *(const v16h*)(V + baseBH + (size_t)(kb + krow) * D + seg);
#pragma unroll
      for (int i = 0; i < 16; ++i) vbufT[(seg + i) * 32 + krow] = vv[i];
    }
#if HAVE_TDM
    if (wave == 0) {
      if (t + 1 < nTiles) {   // kick next tile's DMA, then wait for current only
        tdm_load_2d_f16(Km + baseBH + (size_t)(kb + 32) * D,
                        (unsigned)(uintptr_t)&kbuf[(t + 1) & 1][0], HDIM, 32, (unsigned)D,
                        (unsigned)(S - kb - 32));
        __builtin_amdgcn_s_wait_tensorcnt(1);
      } else {
        __builtin_amdgcn_s_wait_tensorcnt(0);
      }
    }
#else
    {   // fallback: manual K staging
      int tt = threadIdx.x;
      int krow = tt >> 2, seg = (tt & 3) * 16;
      v16h kv = *(const v16h*)(Km + baseBH + (size_t)(kb + krow) * D + seg);
      *(v16h*)&kbuf[t & 1][krow * HDIM + seg] = kv;
    }
#endif
    __syncthreads();
    const _Float16* kcur = &kbuf[t & 1][0];

    // scores: two 16-key tiles, each accumulated over d in two K=32 wmmas
    v8f sc[2];
#pragma unroll
    for (int kt = 0; kt < 2; ++kt) {
      sc[kt] = {};
#pragma unroll
      for (int dk = 0; dk < 2; ++dk) {
        v16h bf = load_b32x16(kcur, kt * 16, HDIM, dk * 32, lane);
        sc[kt] = __builtin_amdgcn_wmma_f32_16x16x32_f16(false, qf[dk], false, bf, (short)0,
                                                        sc[kt], false, false);
      }
      // additive mask: this lane's column = one key
      float madd = (mask[b * S + kb + kt * 16 + col] == 0) ? -1.0e9f : 0.0f;
#pragma unroll
      for (int i = 0; i < 8; ++i) sc[kt][i] += madd;
    }

    // online softmax (row stats reduced across the 16-lane column group)
    float alpha[8], p0[8], p1[8];
#pragma unroll
    for (int i = 0; i < 8; ++i) {
      float v = fmaxf(sc[0][i], sc[1][i]);
      v = fmaxf(v, __shfl_xor(v, 1, 32));
      v = fmaxf(v, __shfl_xor(v, 2, 32));
      v = fmaxf(v, __shfl_xor(v, 4, 32));
      v = fmaxf(v, __shfl_xor(v, 8, 32));
      float mn = fmaxf(mrun[i], v);
      alpha[i] = __expf(mrun[i] - mn);
      mrun[i] = mn;
      p0[i] = __expf(sc[0][i] - mn);
      p1[i] = __expf(sc[1][i] - mn);
      float s = p0[i] + p1[i];
      s += __shfl_xor(s, 1, 32);
      s += __shfl_xor(s, 2, 32);
      s += __shfl_xor(s, 4, 32);
      s += __shfl_xor(s, 8, 32);
      lrun[i] = lrun[i] * alpha[i] + s;
    }
#pragma unroll
    for (int nt = 0; nt < 4; ++nt)
#pragma unroll
      for (int i = 0; i < 8; ++i) acc[nt][i] *= alpha[i];

    // P : C layout -> per-wave LDS tile (16x32 row-major)
#pragma unroll
    for (int i = 0; i < 8; ++i) {
      int r = rsel * 8 + i;
      pbuf[wave][r * 32 + col]      = (_Float16)p0[i];
      pbuf[wave][r * 32 + 16 + col] = (_Float16)p1[i];
    }
    asm volatile("s_wait_dscnt 0" ::: "memory");   // wave-local LDS transpose fence

    // P as A-fragment (16x32), V^T as B-fragments: context += P @ V
    v16h pf = load_a16x32(&pbuf[wave][0], 0, 32, 0, lane);
#pragma unroll
    for (int nt = 0; nt < 4; ++nt) {
      v16h vb = load_b32x16(vbufT, nt * 16, 32, 0, lane);
      acc[nt] = __builtin_amdgcn_wmma_f32_16x16x32_f16(false, pf, false, vb, (short)0,
                                                       acc[nt], false, false);
    }
  }

  // normalize and store context (f16, [B,S,D] with head offset)
#pragma unroll
  for (int i = 0; i < 8; ++i) lrun[i] = (lrun[i] > 0.0f) ? (1.0f / lrun[i]) : 0.0f;
#pragma unroll
  for (int nt = 0; nt < 4; ++nt)
#pragma unroll
    for (int i = 0; i < 8; ++i) {
      int r = q0 + rsel * 8 + i;
      O[baseBH + (size_t)r * D + nt * 16 + col] = (_Float16)(acc[nt][i] * lrun[i]);
    }
}

// ---------------------------------------------------------------------------
extern "C" void kernel_launch(void* const* d_in, const int* in_sizes, int n_in,
                              void* d_out, int out_size, void* d_ws, size_t ws_size,
                              hipStream_t stream) {
  const float* X    = (const float*)d_in[0];
  const int*   mask = (const int*)d_in[1];
  const float* Wq   = (const float*)d_in[2];
  const float* bq   = (const float*)d_in[3];
  const float* Wk   = (const float*)d_in[4];
  const float* bk   = (const float*)d_in[5];
  const float* Wv   = (const float*)d_in[6];
  const float* bv   = (const float*)d_in[7];
  const float* Wo   = (const float*)d_in[8];
  const float* bo   = (const float*)d_in[9];

  const int Bv = 4, Sv = 2048, Dv = 1024;
  const int M = Bv * Sv;                 // 8192
  const size_t MD = (size_t)M * Dv;      // 8388608
  const size_t DD = (size_t)Dv * Dv;     // 1048576

  _Float16* ws  = (_Float16*)d_ws;
  _Float16* Xh  = ws; ws += MD;
  _Float16* Wqh = ws; ws += DD;
  _Float16* Wkh = ws; ws += DD;
  _Float16* Wvh = ws; ws += DD;
  _Float16* Woh = ws; ws += DD;
  _Float16* Qh  = ws; ws += MD;
  _Float16* Kh  = ws; ws += MD;
  _Float16* Vh  = ws; ws += MD;
  _Float16* Ch  = ws; ws += MD;

  // fp32 -> f16 casts
  {
    int threads = 256;
    cast_f32_to_f16<<<(int)((MD / 4 + threads - 1) / threads), threads, 0, stream>>>(X,  Xh,  (int)MD);
    int wb = (int)((DD / 4 + threads - 1) / threads);
    cast_f32_to_f16<<<wb, threads, 0, stream>>>(Wq, Wqh, (int)DD);
    cast_f32_to_f16<<<wb, threads, 0, stream>>>(Wk, Wkh, (int)DD);
    cast_f32_to_f16<<<wb, threads, 0, stream>>>(Wv, Wvh, (int)DD);
    cast_f32_to_f16<<<wb, threads, 0, stream>>>(Wo, Woh, (int)DD);
  }

  // Q/K/V projections (f16 out)
  dim3 gblk(256);
  dim3 gg(M / 256, Dv / 64);
  gemm_bias_wmma<<<gg, gblk, 0, stream>>>(Xh, Wqh, bq, Qh, nullptr, M, Dv, Dv);
  gemm_bias_wmma<<<gg, gblk, 0, stream>>>(Xh, Wkh, bk, Kh, nullptr, M, Dv, Dv);
  gemm_bias_wmma<<<gg, gblk, 0, stream>>>(Xh, Wvh, bv, Vh, nullptr, M, Dv, Dv);

  // attention
  dim3 ag(Sv / 64, Bv * HEADS);
  flash_attn_wmma<<<ag, 128, 0, stream>>>(Qh, Kh, Vh, mask, Ch, Sv, Dv);

  // output projection (f32 out)
  gemm_bias_wmma<<<gg, gblk, 0, stream>>>(Ch, Woh, bo, nullptr, (float*)d_out, M, Dv, Dv);
}
